// SDPA_14577119002968
// MI455X (gfx1250) — matrix operand order repaired and checked by reading, not compiled, not run on hardware
//
#include <hip/hip_runtime.h>

typedef __attribute__((ext_vector_type(16))) _Float16 v16h;
typedef __attribute__((ext_vector_type(8)))  float    v8f;
typedef __attribute__((ext_vector_type(8)))  _Float16 h8;
typedef __attribute__((ext_vector_type(2)))  _Float16 h2;

#define B_  2
#define S_  4096
#define NH_ 16
#define HD_ 64

#define QTILE_WG 128    // q rows per workgroup
#define QTILE_W  16     // q rows per wave
#define KVBLK    32     // kv rows per iteration
#define NWAVES   8
#define NTHREADS 256

#define PK 8            // K LDS row pad (halfs): stride 72 halfs = 144 B (16B-aligned)
#define PP 8            // P LDS row pad (halfs): stride 40 halfs =  80 B (16B-aligned)
#define VBS 24          // V B-frag lane stride (halfs): 48 B (16B-aligned)

static __device__ inline h2 cvt_pk(float a, float b) {
#if __has_builtin(__builtin_amdgcn_cvt_pkrtz)
    return __builtin_bit_cast(h2, __builtin_amdgcn_cvt_pkrtz(a, b)); // v_cvt_pk_rtz_f16_f32
#else
    h2 r; r[0] = (_Float16)a; r[1] = (_Float16)b; return r;
#endif
}

// raw v_exp_f32 (base-2); softmax is computed in the log2 domain
static __device__ inline float exp2_fast(float x) {
#if __has_builtin(__builtin_amdgcn_exp2f)
    return __builtin_amdgcn_exp2f(x);
#else
    return exp2f(x);
#endif
}

// 16-lane-group butterfly exchange of a 32-bit value (VALU pipe when possible)
static __device__ inline unsigned bfly16(unsigned u, unsigned s0, unsigned s1, int off) {
#if __has_builtin(__builtin_amdgcn_permlane16)
    (void)off;
    return __builtin_amdgcn_permlane16(u, u, s0, s1, false, false);
#else
    (void)s0; (void)s1;
    return (unsigned)__shfl_xor((int)u, off, 16);
#endif
}

// packed-f16 max across the 16-lane half-wave group: 2 rows per register
static __device__ inline h2 rowmax16_pk(h2 p) {
#define STEP_(s0, s1, off) p = __builtin_elementwise_max(p, \
        __builtin_bit_cast(h2, bfly16(__builtin_bit_cast(unsigned, p), (s0), (s1), (off))))
    STEP_(0x67452301u, 0xEFCDAB89u, 1);
    STEP_(0x54761032u, 0xDCFE98BAu, 2);
    STEP_(0x32107654u, 0xBA98FEDCu, 4);
    STEP_(0xFEDCBA98u, 0x76543210u, 8);
#undef STEP_
    return p;
}

__global__ __launch_bounds__(NTHREADS)
void sdpa_fwd_wmma(const float* __restrict__ Q,
                   const float* __restrict__ K,
                   const float* __restrict__ V,
                   float* __restrict__ O)
{
    __shared__ __align__(16) _Float16 ldsK[KVBLK][HD_ + PK];
    __shared__ __align__(16) _Float16 ldsVB[4][32][VBS];   // V in PV B-frag layout
    __shared__ __align__(16) _Float16 ldsP[NWAVES][QTILE_W][KVBLK + PP];

    const int tid  = threadIdx.x;
    const int wave = tid >> 5;
    const int lane = tid & 31;
    const int n    = lane & 15;   // column index within a 16-wide tile
    const int hi   = lane >> 4;   // half-wave select (0: lanes 0-15, 1: 16-31)

    const int nQblk = S_ / QTILE_WG;
    const int qblk  = blockIdx.x % nQblk;
    const int bh    = blockIdx.x / nQblk;
    const int h     = bh % NH_;
    const int b     = bh / NH_;

    const int q0 = qblk * QTILE_WG;
    const int qw = q0 + wave * QTILE_W;            // this wave's q-tile base

    const size_t tokStride = (size_t)NH_ * HD_;
    const size_t baseBH    = (size_t)b * S_ * tokStride + (size_t)h * HD_;

    // ---- Q A-fragments; scale folds 1/sqrt(HD) AND log2(e) ---------------
    const float qscale = 0.125f * 1.44269504088896340736f;
    v16h aq0, aq1;
    {
        const int m  = n;                 // A-layout: M = lane & 15
        const int kb = hi * 16;           // hi half starts at K=16
        const float* qrow = Q + baseBH + (size_t)(qw + m) * tokStride;
        #pragma unroll
        for (int e = 0; e < 16; ++e) {
            const int kk = e + kb;
            aq0[e] = (_Float16)(qrow[kk]      * qscale);
            aq1[e] = (_Float16)(qrow[kk + 32] * qscale);
        }
    }

    // ones B-matrix (layout-independent): row sums via WMMA
    v16h ones;
    #pragma unroll
    for (int e = 0; e < 16; ++e) ones[e] = (_Float16)1.0f;

    // ---- online-softmax state (per lane: rows r + hi*8, cols 16d + n) ----
    float rowM[8], rowL[8];
    v8f   acc[4];
    v8f   zero8 = {};
    #pragma unroll
    for (int r = 0; r < 8; ++r) { rowM[r] = -__builtin_inff(); rowL[r] = 0.0f; }
    #pragma unroll
    for (int d = 0; d < 4; ++d) acc[d] = zero8;

    // K staging: 256 threads cover 32 rows x 64 feats, 8 feats per thread
    const int krow  = tid >> 3;          // 0..31
    const int kfeat = (tid & 7) * 8;     // 0,8,...,56
    const size_t kOff = baseBH + (size_t)krow * tokStride + kfeat;

    // V staging: column chunks, stored straight into B-fragment layout.
    const int vf  = tid & 63;            // feature
    const int vkg = tid >> 6;            // kv group of 8 rows
    const int vd  = vf >> 4;                          // d tile
    const int vl  = (vf & 15) + 16 * (vkg >> 1);      // B-frag lane
    const int ve0 = (vkg & 1) * 8;                    // element base (kv & 15)
    const size_t vOff = baseBH + (size_t)(vkg * 8) * tokStride + vf;

    const int kvEnd = q0 + QTILE_WG;     // causal bound; multiple of KVBLK

    for (int c = 0; c < kvEnd; c += KVBLK) {
        // ---- cooperative K/V staging -------------------------------------
        {
            const float* ks = K + kOff + (size_t)c * tokStride;
            float4 k0 = ((const float4*)ks)[0];
            float4 k1 = ((const float4*)ks)[1];
            union { h8 v; h2 p[4]; } uk;
            uk.p[0] = cvt_pk(k0.x, k0.y);  uk.p[1] = cvt_pk(k0.z, k0.w);
            uk.p[2] = cvt_pk(k1.x, k1.y);  uk.p[3] = cvt_pk(k1.z, k1.w);
            *(h8*)&ldsK[krow][kfeat] = uk.v;

            const float* vs = V + vOff + (size_t)c * tokStride;
            float f0 = vs[0 * tokStride], f1 = vs[1 * tokStride];
            float f2 = vs[2 * tokStride], f3 = vs[3 * tokStride];
            float f4 = vs[4 * tokStride], f5 = vs[5 * tokStride];
            float f6 = vs[6 * tokStride], f7 = vs[7 * tokStride];
            union { h8 v; h2 p[4]; } uv;
            uv.p[0] = cvt_pk(f0, f1);  uv.p[1] = cvt_pk(f2, f3);
            uv.p[2] = cvt_pk(f4, f5);  uv.p[3] = cvt_pk(f6, f7);
            *(h8*)&ldsVB[vd][vl][ve0] = uv.v;

            if (c + KVBLK < kvEnd) {       // pull next block L2 -> WGP$
                __builtin_prefetch(ks + (size_t)KVBLK * tokStride, 0, 1);
                __builtin_prefetch(V + kOff + (size_t)(c + KVBLK) * tokStride, 0, 1);
            }
        }
        __syncthreads();

        // wave-uniform causal skip: keeps EXEC all-1s around WMMA
        const bool active = (c <= qw + QTILE_W - 1);
        v16h bvf[4];                       // V fragments, hoisted across barrier
        if (active) {
            // ---- scores S = Qs * K^T, two 16-col kv tiles ----------------
            v8f s0 = zero8, s1 = zero8;
            {
                const int fb = hi * 16;                 // B-layout K offset
                const _Float16* kr0 = &ldsK[n][0];      // kv row c + n
                const _Float16* kr1 = &ldsK[16 + n][0]; // kv row c + 16 + n
                v16h bk00, bk01, bk10, bk11;
                #pragma unroll
                for (int e = 0; e < 16; ++e) bk00[e] = kr0[fb + e];
                #pragma unroll
                for (int e = 0; e < 16; ++e) bk01[e] = kr0[fb + 32 + e];
                #pragma unroll
                for (int e = 0; e < 16; ++e) bk10[e] = kr1[fb + e];
                #pragma unroll
                for (int e = 0; e < 16; ++e) bk11[e] = kr1[fb + 32 + e];
                s0 = __builtin_amdgcn_wmma_f32_16x16x32_f16(false, aq0, false, bk00, (short)0, s0, false, false);
                s0 = __builtin_amdgcn_wmma_f32_16x16x32_f16(false, aq1, false, bk01, (short)0, s0, false, false);
                s1 = __builtin_amdgcn_wmma_f32_16x16x32_f16(false, aq0, false, bk10, (short)0, s1, false, false);
                s1 = __builtin_amdgcn_wmma_f32_16x16x32_f16(false, aq1, false, bk11, (short)0, s1, false, false);
            }
            // hoisted V-fragment loads: fill the WMMA->VALU hazard slots
            #pragma unroll
            for (int d = 0; d < 4; ++d) {
                union { v16h w; h8 g[2]; } ub;
                const h8* pv = (const h8*)&ldsVB[d][lane][0];
                ub.g[0] = pv[0];
                ub.g[1] = pv[1];
                bvf[d] = ub.w;
            }

            // ---- mask only the diagonal block ----------------------------
            float v0a[8], v1a[8];
            const bool needMask = (c + KVBLK - 1 > qw);   // last active block only
            if (needMask) {
                const int j0 = c + n;
                const int j1 = c + 16 + n;
                #pragma unroll
                for (int r = 0; r < 8; ++r) {
                    const int i = qw + r + hi * 8;          // global q row
                    v0a[r] = (j0 <= i) ? s0[r] : -__builtin_inff();
                    v1a[r] = (j1 <= i) ? s1[r] : -__builtin_inff();
                }
            } else {
                #pragma unroll
                for (int r = 0; r < 8; ++r) { v0a[r] = s0[r]; v1a[r] = s1[r]; }
            }

            // ---- row maxima: packed-f16 butterfly (2 rows / register) ----
            h2 pm[4];
            #pragma unroll
            for (int t = 0; t < 4; ++t)
                pm[t] = cvt_pk(fmaxf(v0a[2 * t], v1a[2 * t]),
                               fmaxf(v0a[2 * t + 1], v1a[2 * t + 1]));
            #pragma unroll
            for (int t = 0; t < 4; ++t) pm[t] = rowmax16_pk(pm[t]);

            // ---- online rescale + exp (base-2 domain) --------------------
            #pragma unroll
            for (int r = 0; r < 8; ++r) {
                const float mx   = (float)pm[r >> 1][r & 1];
                const float mNew = fmaxf(rowM[r], mx);
                const float corr = exp2_fast(rowM[r] - mNew);
                rowM[r] = mNew;
                rowL[r] *= corr;                            // sum added below via WMMA
                const float p0 = exp2_fast(v0a[r] - mNew);  // exp2(-inf)=0 masks
                const float p1 = exp2_fast(v1a[r] - mNew);
                #pragma unroll
                for (int d = 0; d < 4; ++d) acc[d][r] *= corr;
                ldsP[wave][r + hi * 8][n]      = (_Float16)p0;
                ldsP[wave][r + hi * 8][16 + n] = (_Float16)p1;
            }
        }
        __syncthreads();   // P visible wave-wide

        if (active) {
            // ---- P as A-fragment (16x32) from LDS ------------------------
            v16h ap;
            {
                const int kb = hi * 16;
                const _Float16* pr = &ldsP[wave][n][0];
                #pragma unroll
                for (int e = 0; e < 16; ++e)
                    ap[e] = pr[e + kb];
            }
            // ---- row sums on the matrix pipe: l += P * Ones --------------
            {
                v8f ssum = __builtin_amdgcn_wmma_f32_16x16x32_f16(false, ap, false, ones, (short)0, zero8, false, false);
                #pragma unroll
                for (int r = 0; r < 8; ++r) rowL[r] += ssum[r];
            }
            // ---- acc += P(16x32) * V(32x16); V fragments pre-loaded ------
            #pragma unroll
            for (int d = 0; d < 4; ++d)
                acc[d] = __builtin_amdgcn_wmma_f32_16x16x32_f16(false, ap, false, bvf[d], (short)0, acc[d], false, false);
        }
        __syncthreads();   // protect K/V LDS before next staging
    }

    // ---- epilogue: normalize by row sums, store fp32 ---------------------
    #pragma unroll
    for (int r = 0; r < 8; ++r) {
        const int   i   = qw + r + hi * 8;
        const float inv = 1.0f / rowL[r];
        float* orow = O + baseBH + (size_t)i * tokStride;
        #pragma unroll
        for (int d = 0; d < 4; ++d)
            orow[d * 16 + n] = acc[d][r] * inv;
    }
}

extern "C" void kernel_launch(void* const* d_in, const int* in_sizes, int n_in,
                              void* d_out, int out_size, void* d_ws, size_t ws_size,
                              hipStream_t stream) {
    (void)in_sizes; (void)n_in; (void)d_ws; (void)ws_size; (void)out_size;
    const float* q = (const float*)d_in[0];
    const float* k = (const float*)d_in[1];
    const float* v = (const float*)d_in[2];
    float*       o = (float*)d_out;
    dim3 grid(B_ * NH_ * (S_ / QTILE_WG));
    dim3 block(NTHREADS);
    hipLaunchKernelGGL(sdpa_fwd_wmma, grid, block, 0, stream, q, k, v, o);
}
